// TransformerEncoder_56057913147406
// MI455X (gfx1250) — matrix operand order repaired and checked by reading, not compiled, not run on hardware
//
#include <hip/hip_runtime.h>
#include <hip/hip_bf16.h>

// ---------------- problem constants ----------------
#define B_   2
#define S_   2048
#define E_   1024
#define A_   64
#define H_   16
#define FF_  4096
#define BS_  (B_ * S_)          // 4096 rows
#define HBSA_ (H_ * BS_ * A_)   // 4,194,304

typedef __bf16 bf16_t;
typedef __attribute__((ext_vector_type(16))) __bf16 v16bf;
typedef __attribute__((ext_vector_type(8)))  __bf16 v8bf;
typedef __attribute__((ext_vector_type(8)))  float  v8f;

// =====================================================================
// WMMA fragment loaders (CDNA5 ISA §7.12.2 layouts, wave32).
// Operands arranged so each lane's 16 bf16 elements are contiguous
// 16-byte runs -> b128 loads.
//
// A 16x32 bf16 (row-major [M x K]): lane L: m = L&15, half = L>>4;
//   elements 0..7  = k in [half*8,    half*8+8)
//   elements 8..15 = k in [16+half*8, 16+half*8+8)
// Bt (B stored transposed, row-major [N x K]): lane L: n = L&15;
//   elements 0..7 = k in [half*8, half*8+8), 8..15 = k in [16+half*8, 16+half*8+8)
// C 16x16 f32: lane L: n = L&15; VGPR v: m = v + 8*(L>>4)
// =====================================================================
__device__ __forceinline__ v16bf combine16(v8bf lo, v8bf hi) {
  return __builtin_shufflevector(lo, hi, 0, 1, 2, 3, 4, 5, 6, 7,
                                 8, 9, 10, 11, 12, 13, 14, 15);
}

__device__ __forceinline__ v16bf load_a_frag(const bf16_t* __restrict__ A,
                                             int lda, int mBase, int kBase, int lane) {
  const bf16_t* p = A + (size_t)(mBase + (lane & 15)) * lda + kBase + ((lane >> 4) << 3);
  v8bf lo = *(const v8bf*)p;         // k: half*8 .. +7
  v8bf hi = *(const v8bf*)(p + 16);  // k: 16+half*8 .. +7
  return combine16(lo, hi);
}

// Bt row-major [N x K]
__device__ __forceinline__ v16bf load_bt_frag(const bf16_t* __restrict__ Bt,
                                              int ldbt, int kBase, int nBase, int lane) {
  const bf16_t* p = Bt + (size_t)(nBase + (lane & 15)) * ldbt + kBase + ((lane >> 4) << 3);
  v8bf lo = *(const v8bf*)p;         // k: half*8 .. +7
  v8bf hi = *(const v8bf*)(p + 16);  // k: 16+half*8 .. +7
  return combine16(lo, hi);
}

__device__ __forceinline__ v8f wmma_bf16(v16bf a, v16bf b, v8f c) {
  return __builtin_amdgcn_wmma_f32_16x16x32_bf16(false, a, false, b, (short)0, c,
                                                 false, false);
}

// =====================================================================
// Software-pipelined wave-level GEMM tile: 16 rows x 64 cols, K steps of
// 64 with ping-pong fragment buffers so WMMAs consume fragments whose
// loads were issued one full stage earlier (no wait-on-own-load stall).
// Requires K % 64 == 0, K >= 64.
// =====================================================================
__device__ __forceinline__ void wave_gemm_16x64(
    const bf16_t* __restrict__ A, int lda,
    const bf16_t* __restrict__ Bt, int ldbt,
    int mBase, int nBase, int K, int lane, v8f acc[4]) {
  // prologue: buffer 0 <- k = 0
  v16bf a0 = load_a_frag(A, lda, mBase, 0, lane);
  v16bf b0_0 = load_bt_frag(Bt, ldbt, 0, nBase + 0,  lane);
  v16bf b0_1 = load_bt_frag(Bt, ldbt, 0, nBase + 16, lane);
  v16bf b0_2 = load_bt_frag(Bt, ldbt, 0, nBase + 32, lane);
  v16bf b0_3 = load_bt_frag(Bt, ldbt, 0, nBase + 48, lane);

  for (int k = 0; k < K; k += 64) {
    // stage 1: issue loads for k+32 (buffer 1), compute on buffer 0
    v16bf a1 = load_a_frag(A, lda, mBase, k + 32, lane);
    v16bf b1_0 = load_bt_frag(Bt, ldbt, k + 32, nBase + 0,  lane);
    v16bf b1_1 = load_bt_frag(Bt, ldbt, k + 32, nBase + 16, lane);
    v16bf b1_2 = load_bt_frag(Bt, ldbt, k + 32, nBase + 32, lane);
    v16bf b1_3 = load_bt_frag(Bt, ldbt, k + 32, nBase + 48, lane);
    if (k + 64 < K)  // stream next A strip toward L2 (global_prefetch_b8)
      __builtin_prefetch(A + (size_t)(mBase + (lane & 15)) * lda + k + 64, 0, 1);
    acc[0] = wmma_bf16(a0, b0_0, acc[0]);
    acc[1] = wmma_bf16(a0, b0_1, acc[1]);
    acc[2] = wmma_bf16(a0, b0_2, acc[2]);
    acc[3] = wmma_bf16(a0, b0_3, acc[3]);

    // stage 2: issue loads for k+64 (buffer 0), compute on buffer 1
    if (k + 64 < K) {
      a0 = load_a_frag(A, lda, mBase, k + 64, lane);
      b0_0 = load_bt_frag(Bt, ldbt, k + 64, nBase + 0,  lane);
      b0_1 = load_bt_frag(Bt, ldbt, k + 64, nBase + 16, lane);
      b0_2 = load_bt_frag(Bt, ldbt, k + 64, nBase + 32, lane);
      b0_3 = load_bt_frag(Bt, ldbt, k + 64, nBase + 48, lane);
    }
    acc[0] = wmma_bf16(a1, b1_0, acc[0]);
    acc[1] = wmma_bf16(a1, b1_1, acc[1]);
    acc[2] = wmma_bf16(a1, b1_2, acc[2]);
    acc[3] = wmma_bf16(a1, b1_3, acc[3]);
  }
}

// =====================================================================
// Elementwise / layout helpers
// =====================================================================
__global__ __launch_bounds__(256) void add_cvt_kernel(const float* __restrict__ a,
                                                      const float* __restrict__ b,
                                                      float* __restrict__ xf,
                                                      bf16_t* __restrict__ xb, int n) {
  int i = blockIdx.x * 256 + threadIdx.x;
  if (i < n) {
    float v = a[i] + b[i];
    xf[i] = v;
    xb[i] = (bf16_t)v;
  }
}

// fp32 [R x C] row-major -> bf16 [C x R] (one-time weight pre-transpose so
// GEMM hot loops read contiguous-K b128 fragments). blockIdx.y batches.
__global__ __launch_bounds__(256) void cvt_t_kernel(const float* __restrict__ a,
                                                    bf16_t* __restrict__ o,
                                                    int R, int C) {
  const size_t mat = (size_t)blockIdx.y * R * C;
  int i = blockIdx.x * 256 + threadIdx.x;
  if (i < R * C) {
    int r = i / C, c = i % C;
    o[mat + (size_t)c * R + r] = (bf16_t)a[mat + i];
  }
}

// =====================================================================
// Generic GEMM: C[M,N] = A[M,K](bf16,row) * Bt[N,K]^T (+bias)(+relu)
// 256 thr = 8 waves; wave w -> rows [blockIdx.y*128 + w*16), cols
// [blockIdx.x*64).
// =====================================================================
__global__ __launch_bounds__(256) void gemm_bf16_kernel(
    const bf16_t* __restrict__ A, const bf16_t* __restrict__ Bt,
    int M, int N, int K, const float* __restrict__ bias, int relu,
    float* __restrict__ Cf, bf16_t* __restrict__ Cb) {
  const int lane = threadIdx.x & 31;
  const int w    = threadIdx.x >> 5;
  const int mBase = blockIdx.y * 128 + w * 16;
  const int nBase = blockIdx.x * 64;
  if (mBase >= M) return;

  v8f acc[4];
#pragma unroll
  for (int i = 0; i < 4; ++i)
#pragma unroll
    for (int j = 0; j < 8; ++j) acc[i][j] = 0.f;

  wave_gemm_16x64(A, K, Bt, K, mBase, nBase, K, lane, acc);

  const int n0 = lane & 15;
  const int mOff = (lane >> 4) * 8;
#pragma unroll
  for (int nt = 0; nt < 4; ++nt)
#pragma unroll
    for (int v = 0; v < 8; ++v) {
      const int row = mBase + mOff + v;
      const int col = nBase + nt * 16 + n0;
      float val = acc[nt][v];
      if (bias) val += bias[col];
      if (relu) val = fmaxf(val, 0.f);
      const size_t idx = (size_t)row * N + col;
      if (Cf) Cf[idx] = val;
      if (Cb) Cb[idx] = (bf16_t)val;
    }
}

// =====================================================================
// QKV GEMM: z = which*H + h ; per head [BS x E] x (Wt[A x E])^T.
// Q,K row-major bf16 (row-major K IS the Bt form for Q*K^T); V written
// TRANSPOSED bf16 [H][A][BS] for contiguous-K P*V fragments.
// fp32 K/V go straight into d_out.
// =====================================================================
__global__ __launch_bounds__(256) void qkv_gemm_kernel(
    const bf16_t* __restrict__ xb, const bf16_t* __restrict__ wqkvT,
    bf16_t* __restrict__ qkb, bf16_t* __restrict__ vtb,
    float* __restrict__ kout, float* __restrict__ vout) {
  const int lane = threadIdx.x & 31;
  const int w    = threadIdx.x >> 5;
  const int z     = blockIdx.z;        // 0..3H-1
  const int which = z >> 4;            // /H_
  const int h     = z & (H_ - 1);
  const int mBase = blockIdx.y * 128 + w * 16;
  const bf16_t* Bt = wqkvT + (size_t)z * (E_ * A_);   // [A x E] row-major

  v8f acc[4];
#pragma unroll
  for (int i = 0; i < 4; ++i)
#pragma unroll
    for (int j = 0; j < 8; ++j) acc[i][j] = 0.f;

  wave_gemm_16x64(xb, E_, Bt, E_, mBase, 0, E_, lane, acc);

  const int n0 = lane & 15;
  const int mOff = (lane >> 4) * 8;
  float* fdst = (which == 1) ? (kout + (size_t)h * BS_ * A_)
              : (which == 2) ? (vout + (size_t)h * BS_ * A_) : nullptr;
#pragma unroll
  for (int nt = 0; nt < 4; ++nt)
#pragma unroll
    for (int v = 0; v < 8; ++v) {
      const int row = mBase + mOff + v;
      const int col = nt * 16 + n0;
      const float val = acc[nt][v];
      if (which < 2) {
        qkb[(size_t)z * BS_ * A_ + (size_t)row * A_ + col] = (bf16_t)val;
      } else {
        vtb[(size_t)h * A_ * BS_ + (size_t)col * BS_ + row] = (bf16_t)val;
      }
      if (fdst) fdst[(size_t)row * A_ + col] = val;
    }
}

// =====================================================================
// Flash attention, causal, scale = 1/sqrt(E) = 1/32 (reference quirk).
// grid (S/128, B, H); 8 waves/block; wave owns one 16-row Q tile,
// streams 32-key chunks. Pipelining: after the QK^T WMMAs, the V
// fragments for this chunk and the K fragments for the NEXT chunk are
// issued immediately, hiding their latency behind the softmax VALU work.
// =====================================================================
__global__ __launch_bounds__(256) void attention_kernel(
    const bf16_t* __restrict__ qkb, const bf16_t* __restrict__ vtb,
    bf16_t* __restrict__ attnb) {
  __shared__ __attribute__((aligned(16))) bf16_t psh[8][16][32];

  const int lane = threadIdx.x & 31;
  const int w    = threadIdx.x >> 5;
  const int hf   = lane >> 4;
  const int n0   = lane & 15;
  const int h = blockIdx.z;
  const int b = blockIdx.y;
  const int mBase = blockIdx.x * 128 + w * 16;

  const bf16_t* Qh = qkb + (size_t)(0 * H_ + h) * BS_ * A_;  // [BS x A]
  const bf16_t* Kh = qkb + (size_t)(1 * H_ + h) * BS_ * A_;  // [BS x A] == Bt form
  const bf16_t* Vt = vtb + (size_t)h * A_ * BS_;             // [A x BS] == Bt form
  const int rowQ = b * S_ + mBase;

  const v16bf aq0 = load_a_frag(Qh, A_, rowQ, 0, lane);
  const v16bf aq1 = load_a_frag(Qh, A_, rowQ, 32, lane);

  const float NEG_INF = -__builtin_inff();
  float mrun[8], lrun[8];
  v8f accO[4];
#pragma unroll
  for (int v = 0; v < 8; ++v) { mrun[v] = NEG_INF; lrun[v] = 0.f; }
#pragma unroll
  for (int i = 0; i < 4; ++i)
#pragma unroll
    for (int j = 0; j < 8; ++j) accO[i][j] = 0.f;

  const float inv_scale = 1.0f / 32.0f;  // 1/sqrt(E_)
  const int nChunks = mBase / 32 + 1;    // causal: keys up to mBase+15

  // preload K fragments for chunk 0
  v16bf bk[2][2];
#pragma unroll
  for (int j = 0; j < 2; ++j) {
    bk[j][0] = load_bt_frag(Kh, A_, 0,  b * S_ + j * 16, lane);
    bk[j][1] = load_bt_frag(Kh, A_, 32, b * S_ + j * 16, lane);
  }

  for (int c = 0; c < nChunks; ++c) {
    const int kc = c * 32;

    // ---- scores: two 16x16 tiles over keys [kc, kc+32) ----
    v8f sc[2];
#pragma unroll
    for (int j = 0; j < 2; ++j) {
      v8f t;
#pragma unroll
      for (int q = 0; q < 8; ++q) t[q] = 0.f;
      t = wmma_bf16(aq0, bk[j][0], t);
      t = wmma_bf16(aq1, bk[j][1], t);
      sc[j] = t;
    }

    // ---- early-issue: V frags (this chunk) + K frags (next chunk) ----
    v16bf bv0 = load_bt_frag(Vt, BS_, b * S_ + kc, 0,  lane);
    v16bf bv1 = load_bt_frag(Vt, BS_, b * S_ + kc, 16, lane);
    v16bf bv2 = load_bt_frag(Vt, BS_, b * S_ + kc, 32, lane);
    v16bf bv3 = load_bt_frag(Vt, BS_, b * S_ + kc, 48, lane);
    if (c + 1 < nChunks) {
#pragma unroll
      for (int j = 0; j < 2; ++j) {
        bk[j][0] = load_bt_frag(Kh, A_, 0,  b * S_ + kc + 32 + j * 16, lane);
        bk[j][1] = load_bt_frag(Kh, A_, 32, b * S_ + kc + 32 + j * 16, lane);
      }
    }

    // ---- online softmax update (rows live in 16-lane groups) ----
    float p0[8], p1[8];
#pragma unroll
    for (int v = 0; v < 8; ++v) {
      const int srow = mBase + hf * 8 + v;
      float s0 = sc[0][v] * inv_scale;
      float s1 = sc[1][v] * inv_scale;
      if (kc + 0 * 16 + n0 > srow) s0 = NEG_INF;
      if (kc + 1 * 16 + n0 > srow) s1 = NEG_INF;

      float r = fmaxf(s0, s1);
      r = fmaxf(r, __shfl_xor(r, 1, 32));
      r = fmaxf(r, __shfl_xor(r, 2, 32));
      r = fmaxf(r, __shfl_xor(r, 4, 32));
      r = fmaxf(r, __shfl_xor(r, 8, 32));      // stays inside 16-lane group
      const float mnew = fmaxf(mrun[v], r);
      const float corr = __expf(mrun[v] - mnew);
      mrun[v] = mnew;

      const float e0 = __expf(s0 - mnew);
      const float e1 = __expf(s1 - mnew);
      float rs = e0 + e1;
      rs += __shfl_xor(rs, 1, 32);
      rs += __shfl_xor(rs, 2, 32);
      rs += __shfl_xor(rs, 4, 32);
      rs += __shfl_xor(rs, 8, 32);
      lrun[v] = lrun[v] * corr + rs;
#pragma unroll
      for (int nt = 0; nt < 4; ++nt) accO[nt][v] *= corr;
      p0[v] = e0;
      p1[v] = e1;
    }

    // ---- stage P (C-layout) into LDS, reload as b128 A-fragment ----
#pragma unroll
    for (int v = 0; v < 8; ++v) {
      psh[w][hf * 8 + v][0 * 16 + n0] = (bf16_t)p0[v];
      psh[w][hf * 8 + v][1 * 16 + n0] = (bf16_t)p1[v];
    }
    asm volatile("s_wait_dscnt 0" ::: "memory");  // DS in-order within wave

    v16bf pa;
    {
      const bf16_t* pp = &psh[w][lane & 15][hf << 3];
      v8bf lo = *(const v8bf*)pp;         // k: half*8 .. +7
      v8bf hi = *(const v8bf*)(pp + 16);  // k: 16+half*8 .. +7
      pa = combine16(lo, hi);
    }
    asm volatile("s_wait_dscnt 0" ::: "memory");  // reads done before overwrite

    // ---- accO += P[16x32] * V[32x64] (V stored transposed) ----
    accO[0] = wmma_bf16(pa, bv0, accO[0]);
    accO[1] = wmma_bf16(pa, bv1, accO[1]);
    accO[2] = wmma_bf16(pa, bv2, accO[2]);
    accO[3] = wmma_bf16(pa, bv3, accO[3]);
  }

  // ---- normalize, write concat-head output (bf16, feeds proj GEMM) ----
#pragma unroll
  for (int nt = 0; nt < 4; ++nt)
#pragma unroll
    for (int v = 0; v < 8; ++v) {
      const int row = b * S_ + mBase + hf * 8 + v;
      const int col = h * A_ + nt * 16 + n0;
      attnb[(size_t)row * E_ + col] = (bf16_t)(accO[nt][v] / lrun[v]);
    }
}

// =====================================================================
// LayerNorm kernels (row = 1024 cols, block = 256 threads x 4 cols)
// =====================================================================
__device__ __forceinline__ float block_sum(float v, float* red) {
  const int t = threadIdx.x;
  red[t] = v;
  __syncthreads();
  for (int s = 128; s > 0; s >>= 1) {
    if (t < s) red[t] += red[t + s];
    __syncthreads();
  }
  const float r = red[0];
  __syncthreads();
  return r;
}

__global__ __launch_bounds__(256) void ln_add_kernel(
    const float* __restrict__ X, const float* __restrict__ Y,
    const float* __restrict__ g, const float* __restrict__ be,
    float* __restrict__ of, bf16_t* __restrict__ ob) {
  __shared__ float red[256];
  const int row = blockIdx.x, t = threadIdx.x;
  const float* xr = X + (size_t)row * E_;
  const float* yr = Y + (size_t)row * E_;
  float v4[4];
  float s = 0.f;
#pragma unroll
  for (int i = 0; i < 4; ++i) { v4[i] = xr[t + i * 256] + yr[t + i * 256]; s += v4[i]; }
  const float mu = block_sum(s, red) * (1.f / E_);
  float vs = 0.f;
#pragma unroll
  for (int i = 0; i < 4; ++i) { const float d = v4[i] - mu; vs += d * d; }
  const float rstd = rsqrtf(block_sum(vs, red) * (1.f / E_) + 1e-5f);
#pragma unroll
  for (int i = 0; i < 4; ++i) {
    const int col = t + i * 256;
    const float o = (v4[i] - mu) * rstd * g[col] + be[col];
    const size_t idx = (size_t)row * E_ + col;
    if (of) of[idx] = o;
    if (ob) ob[idx] = (bf16_t)o;
  }
}

// fused: ff_out = LN(mha+ff; g1,b1); encoded = LN(mha+ff_out; g2,b2)
__global__ __launch_bounds__(256) void ln2_kernel(
    const float* __restrict__ mha, const float* __restrict__ ff,
    const float* __restrict__ g1, const float* __restrict__ be1,
    const float* __restrict__ g2, const float* __restrict__ be2,
    float* __restrict__ enc) {
  __shared__ float red[256];
  const int row = blockIdx.x, t = threadIdx.x;
  const float* mr = mha + (size_t)row * E_;
  const float* fr = ff  + (size_t)row * E_;
  float m4[4], t4[4];
  float s = 0.f;
#pragma unroll
  for (int i = 0; i < 4; ++i) {
    m4[i] = mr[t + i * 256];
    t4[i] = m4[i] + fr[t + i * 256];
    s += t4[i];
  }
  const float mu = block_sum(s, red) * (1.f / E_);
  float vs = 0.f;
#pragma unroll
  for (int i = 0; i < 4; ++i) { const float d = t4[i] - mu; vs += d * d; }
  const float rstd = rsqrtf(block_sum(vs, red) * (1.f / E_) + 1e-5f);

  float u4[4];
  float s2 = 0.f;
#pragma unroll
  for (int i = 0; i < 4; ++i) {
    const int col = t + i * 256;
    const float ffo = (t4[i] - mu) * rstd * g1[col] + be1[col];
    u4[i] = m4[i] + ffo;
    s2 += u4[i];
  }
  const float mu2 = block_sum(s2, red) * (1.f / E_);
  float vs2 = 0.f;
#pragma unroll
  for (int i = 0; i < 4; ++i) { const float d = u4[i] - mu2; vs2 += d * d; }
  const float rstd2 = rsqrtf(block_sum(vs2, red) * (1.f / E_) + 1e-5f);
#pragma unroll
  for (int i = 0; i < 4; ++i) {
    const int col = t + i * 256;
    enc[(size_t)row * E_ + col] = (u4[i] - mu2) * rstd2 * g2[col] + be2[col];
  }
}

// =====================================================================
// Host launcher
// =====================================================================
extern "C" void kernel_launch(void* const* d_in, const int* in_sizes, int n_in,
                              void* d_out, int out_size, void* d_ws, size_t ws_size,
                              hipStream_t stream) {
  (void)in_sizes; (void)n_in; (void)out_size; (void)ws_size;

  const float* emb    = (const float*)d_in[0];
  const float* pos    = (const float*)d_in[1];
  const float* Wq     = (const float*)d_in[2];
  const float* Wk     = (const float*)d_in[3];
  const float* Wv     = (const float*)d_in[4];
  const float* Wproj  = (const float*)d_in[5];
  const float* W1     = (const float*)d_in[6];
  const float* b1     = (const float*)d_in[7];
  const float* W2     = (const float*)d_in[8];
  const float* b2     = (const float*)d_in[9];
  const float* g_attn = (const float*)d_in[10];
  const float* be_attn= (const float*)d_in[11];
  const float* g_ffn  = (const float*)d_in[12];
  const float* be_ffn = (const float*)d_in[13];
  const float* g_out  = (const float*)d_in[14];
  const float* be_out = (const float*)d_in[15];

  float* out = (float*)d_out;                 // [encoded | K | V]
  float* Kout = out + (size_t)BS_ * E_;
  float* Vout = Kout + (size_t)HBSA_;

  // ---- carve workspace ----
  char* ws = (char*)d_ws;
  size_t off = 0;
  auto carve = [&](size_t bytes) -> char* {
    char* p = ws + off;
    off += (bytes + 255) & ~(size_t)255;
    return p;
  };
  float*  xf     = (float*) carve((size_t)BS_ * E_ * 4);
  bf16_t* xb     = (bf16_t*)carve((size_t)BS_ * E_ * 2);
  bf16_t* wqkvT  = (bf16_t*)carve((size_t)3 * H_ * E_ * A_ * 2);  // [3H][A x E]
  bf16_t* wprojT = (bf16_t*)carve((size_t)E_ * E_ * 2);           // [E x E]
  bf16_t* w1T    = (bf16_t*)carve((size_t)E_ * FF_ * 2);          // [FF x E]
  bf16_t* w2T    = (bf16_t*)carve((size_t)FF_ * E_ * 2);          // [E x FF]
  bf16_t* qkb    = (bf16_t*)carve((size_t)2 * HBSA_ * 2);         // Q,K row-major
  bf16_t* vtb    = (bf16_t*)carve((size_t)HBSA_ * 2);             // V transposed
  bf16_t* attnb  = (bf16_t*)carve((size_t)BS_ * E_ * 2);
  float*  projf  = (float*) carve((size_t)BS_ * E_ * 4);
  float*  mhaf   = (float*) carve((size_t)BS_ * E_ * 4);
  bf16_t* mhab   = (bf16_t*)carve((size_t)BS_ * E_ * 2);
  bf16_t* hidb   = (bf16_t*)carve((size_t)BS_ * FF_ * 2);
  float*  fff    = (float*) carve((size_t)BS_ * E_ * 4);

  const int nXE = BS_ * E_;
  const int nEA = E_ * A_;

  // 1) x = emb + pos (fp32 residual + bf16 operand)
  add_cvt_kernel<<<nXE / 256, 256, 0, stream>>>(emb, pos, xf, xb, nXE);

  // 2) weight transpose-converts (one-time)
  cvt_t_kernel<<<dim3(nEA / 256, H_), 256, 0, stream>>>(Wq, wqkvT,               E_, A_);
  cvt_t_kernel<<<dim3(nEA / 256, H_), 256, 0, stream>>>(Wk, wqkvT + (size_t)H_ * nEA,     E_, A_);
  cvt_t_kernel<<<dim3(nEA / 256, H_), 256, 0, stream>>>(Wv, wqkvT + (size_t)2 * H_ * nEA, E_, A_);
  cvt_t_kernel<<<dim3((E_ * E_) / 256, 1), 256, 0, stream>>>(Wproj, wprojT, E_, E_);
  cvt_t_kernel<<<dim3((E_ * FF_) / 256, 1), 256, 0, stream>>>(W1, w1T, E_, FF_);
  cvt_t_kernel<<<dim3((FF_ * E_) / 256, 1), 256, 0, stream>>>(W2, w2T, FF_, E_);

  // 3) QKV projections (48 head-matrices); fp32 K/V direct to d_out
  qkv_gemm_kernel<<<dim3(1, BS_ / 128, 3 * H_), 256, 0, stream>>>(
      xb, wqkvT, qkb, vtb, Kout, Vout);

  // 4) causal flash attention -> concat-head bf16 [BS, E]
  attention_kernel<<<dim3(S_ / 128, B_, H_), 256, 0, stream>>>(qkb, vtb, attnb);

  // 5) output projection (no bias)
  gemm_bf16_kernel<<<dim3(E_ / 64, BS_ / 128), 256, 0, stream>>>(
      attnb, wprojT, BS_, E_, E_, nullptr, 0, projf, nullptr);

  // 6) mha = LN(x + proj)
  ln_add_kernel<<<BS_, 256, 0, stream>>>(xf, projf, g_attn, be_attn, mhaf, mhab);

  // 7) hidden = mha @ W1 + b1 (no activation)
  gemm_bf16_kernel<<<dim3(FF_ / 64, BS_ / 128), 256, 0, stream>>>(
      mhab, w1T, BS_, FF_, E_, b1, 0, nullptr, hidb);

  // 8) ff = relu(hidden @ W2 + b2)
  gemm_bf16_kernel<<<dim3(E_ / 64, BS_ / 128), 256, 0, stream>>>(
      hidb, w2T, BS_, E_, FF_, b2, 1, fff, nullptr);

  // 9) ff_out = LN(mha+ff); encoded = LN(mha+ff_out) -> d_out
  ln2_kernel<<<BS_, 256, 0, stream>>>(mhaf, fff, g_ffn, be_ffn, g_out, be_out, out);
}
